// VectorQuantizer_62311385530482
// MI455X (gfx1250) — matrix operand order repaired and checked by reading, not compiled, not run on hardware
//
#include <hip/hip_runtime.h>

// Problem constants (from the reference): N=32768 points, K=8192 codes, D=256.
#define NPTS   32768
#define KCODES 8192
#define DIM    256
#define MBLK   64                 // rows of x per block
#define AS     264                // padded LDS row stride in ushorts (16B aligned, bank-conflict-free)
#define NDOUT  (NPTS * DIM)
#define NBLKS  (NPTS / MBLK)      // 512 main blocks
#define JTILES (KCODES / 32)      // 256 column-tile iterations of 32 codes

typedef __attribute__((ext_vector_type(16))) __bf16 v16bf;
typedef __attribute__((ext_vector_type(8)))  __bf16 v8bf;
typedef __attribute__((ext_vector_type(8)))  float  v8f;

#define CAT16(a, b) __builtin_shufflevector(a, b, 0,1,2,3,4,5,6,7,8,9,10,11,12,13,14,15)

__device__ __forceinline__ unsigned short f2bf(float f) {
  unsigned u = __float_as_uint(f);
  u += 0x7FFFu + ((u >> 16) & 1u);          // round-to-nearest-even
  return (unsigned short)(u >> 16);
}
__device__ __forceinline__ float bf2f(unsigned short h) {
  return __uint_as_float(((unsigned)h) << 16);
}

// ---------------------------------------------------------------------------
// Kernel 1: enorm[k] = 0.5 * ||e_k||^2 (exact fp32), one wave per code row.
// ---------------------------------------------------------------------------
__global__ __launch_bounds__(256) void vq_enorm_kernel(const float* __restrict__ emb,
                                                       float* __restrict__ enorm) {
  const int w = threadIdx.x >> 5, lane = threadIdx.x & 31;
  const int row = blockIdx.x * 8 + w;
  const float* ep = emb + (size_t)row * DIM;
  float s = 0.f;
#pragma unroll
  for (int i = 0; i < 8; ++i) {
    float v = ep[lane + (i << 5)];
    s += v * v;
  }
#pragma unroll
  for (int off = 16; off; off >>= 1) s += __shfl_xor(s, off, 32);
  if (lane == 0) enorm[row] = 0.5f * s;
}

// ---------------------------------------------------------------------------
// Kernel 2: main GEMM (bf16 hi/lo x3 WMMA, fp32 accumulate) + argmax + gather
// + per-block partial MSE. Block = 256 threads = 8 waves (4 M-tiles x 2 N-tiles).
// ---------------------------------------------------------------------------
__global__ __launch_bounds__(256) void vq_main_kernel(const float* __restrict__ x,
                                                      const float* __restrict__ emb,
                                                      const float* __restrict__ enorm,
                                                      float* __restrict__ outq,
                                                      float* __restrict__ outmap,
                                                      float* __restrict__ partial) {
  __shared__ __align__(16) unsigned short Ahi[MBLK][AS];  // 33792 B
  __shared__ __align__(16) unsigned short Alo[MBLK][AS];  // 33792 B
  __shared__ __align__(16) unsigned short Bhi[32][AS];    // 16896 B
  __shared__ __align__(16) unsigned short Blo[32][AS];    // 16896 B
  __shared__ int   bestk[MBLK];
  __shared__ float red2[256];

  const int t   = threadIdx.x;
  const int blk = blockIdx.x;
  const int w = t >> 5, lane = t & 31;
  const int wm = w >> 1, wn = w & 1;
  const int half = lane >> 4, l16 = lane & 15;
  const int arow = wm * 16 + l16;       // A fragment row (M = lane%16)
  const int brow = wn * 16 + l16;       // B fragment column within 32-tile

  // ---- stage A: x rows [blk*64, +64) as bf16 hi/lo planes ----
  {
    const int row = t >> 2, q = t & 3;
    const float4* xp = (const float4*)(x + (size_t)(blk * MBLK + row) * DIM);
#pragma unroll
    for (int i = 0; i < 16; ++i) {
      const int c4 = q + (i << 2);            // 0..63
      float4 v = xp[c4];
      const int d = c4 << 2;
      unsigned short h;
      h = f2bf(v.x); Ahi[row][d + 0] = h; Alo[row][d + 0] = f2bf(v.x - bf2f(h));
      h = f2bf(v.y); Ahi[row][d + 1] = h; Alo[row][d + 1] = f2bf(v.y - bf2f(h));
      h = f2bf(v.z); Ahi[row][d + 2] = h; Alo[row][d + 2] = f2bf(v.z - bf2f(h));
      h = f2bf(v.w); Ahi[row][d + 3] = h; Alo[row][d + 3] = f2bf(v.w - bf2f(h));
    }
  }

  float best[8];
  int   bidx[8];
#pragma unroll
  for (int r = 0; r < 8; ++r) { best[r] = -3.4e38f; bidx[r] = 0; }

  for (int jb = 0; jb < JTILES; ++jb) {
    __syncthreads();  // previous tile's B fully consumed (also covers A staging on jb==0)

    // ---- stage B: 32 code rows as bf16 hi/lo (B[k][n] = e[n][k] -> per-column K-contiguous) ----
    {
      const int n = t >> 3, q = t & 7;
      const float4* ep = (const float4*)(emb + (size_t)(jb * 32 + n) * DIM);
#pragma unroll
      for (int i = 0; i < 8; ++i) {
        const int c4 = q + (i << 3);          // 0..63
        float4 v = ep[c4];
        const int d = c4 << 2;
        unsigned short h;
        h = f2bf(v.x); Bhi[n][d + 0] = h; Blo[n][d + 0] = f2bf(v.x - bf2f(h));
        h = f2bf(v.y); Bhi[n][d + 1] = h; Blo[n][d + 1] = f2bf(v.y - bf2f(h));
        h = f2bf(v.z); Bhi[n][d + 2] = h; Blo[n][d + 2] = f2bf(v.z - bf2f(h));
        h = f2bf(v.w); Bhi[n][d + 3] = h; Blo[n][d + 3] = f2bf(v.w - bf2f(h));
      }
    }
    if (jb + 1 < JTILES)  // pull next tile toward L2/L0 (global_prefetch_b8)
      __builtin_prefetch(emb + (size_t)(jb + 1) * 32 * DIM + (size_t)t * 32, 0, 0);
    __syncthreads();

    // ---- 16x16x256 fp32-accurate dot via bf16x3 WMMA ----
    v8f acc = {};
#pragma unroll
    for (int k0 = 0; k0 < DIM; k0 += 32) {
      // A 16x32 bf16 layout: VGPR0-3 <- K = k0+8*half+[0..7]; VGPR4-7 <- +16
      v8bf ah0 = *(const v8bf*)&Ahi[arow][k0 + 8 * half];
      v8bf ah1 = *(const v8bf*)&Ahi[arow][k0 + 16 + 8 * half];
      v8bf al0 = *(const v8bf*)&Alo[arow][k0 + 8 * half];
      v8bf al1 = *(const v8bf*)&Alo[arow][k0 + 16 + 8 * half];
      // B 32x16 bf16 layout: VGPR0-3 <- K = k0+8*half+[0..7]; VGPR4-7 <- +16
      v8bf bh0 = *(const v8bf*)&Bhi[brow][k0 + 8 * half];
      v8bf bh1 = *(const v8bf*)&Bhi[brow][k0 + 16 + 8 * half];
      v8bf bl0 = *(const v8bf*)&Blo[brow][k0 + 8 * half];
      v8bf bl1 = *(const v8bf*)&Blo[brow][k0 + 16 + 8 * half];
      v16bf AH = CAT16(ah0, ah1), AL = CAT16(al0, al1);
      v16bf BH = CAT16(bh0, bh1), BL = CAT16(bl0, bl1);
      acc = __builtin_amdgcn_wmma_f32_16x16x32_bf16(false, AH, false, BH, (short)0, acc, false, false);
      acc = __builtin_amdgcn_wmma_f32_16x16x32_bf16(false, AH, false, BL, (short)0, acc, false, false);
      acc = __builtin_amdgcn_wmma_f32_16x16x32_bf16(false, AL, false, BH, (short)0, acc, false, false);
    }

    // score = x.e - 0.5||e||^2 ; argmin(d2) == argmax(score). C layout: acc[r] <-> M=r+8*half, N=l16.
    const int col = jb * 32 + wn * 16 + l16;
    const float en = enorm[col];
#pragma unroll
    for (int r = 0; r < 8; ++r) {
      float s = acc[r] - en;
      if (s > best[r]) { best[r] = s; bidx[r] = col; }   // strict > keeps first (lowest) col
    }
  }

  // ---- cross-lane argmax reduction: 32 candidates per row (16 residues x 2 wn) ----
  __syncthreads();
  float* redv = (float*)&Ahi[0][0];   // 2048 floats (A/B no longer needed)
  int*   redi = (int*)&Alo[0][0];     // 2048 ints
#pragma unroll
  for (int r = 0; r < 8; ++r) {
    const int rl = wm * 16 + 8 * half + r;
    const int slot = rl * 32 + wn * 16 + l16;
    redv[slot] = best[r];
    redi[slot] = bidx[r];
  }
  __syncthreads();
  if (t < MBLK) {
    float bv = -3.4e38f; int bi = 0;
    for (int i = 0; i < 32; ++i) {
      float v = redv[t * 32 + i]; int ix = redi[t * 32 + i];
      if (v > bv || (v == bv && ix < bi)) { bv = v; bi = ix; }
    }
    bestk[t] = bi;
    outmap[blk * MBLK + t] = (float)bi;   // mapping_inds as float in d_out
  }
  __syncthreads();

  // ---- gather e[k*] -> quantized output; accumulate squared error ----
  float lsum = 0.f;
  {
    const int row = t >> 2, q = t & 3;
    const int k = bestk[row];
    const size_t rbase = (size_t)(blk * MBLK + row) * DIM;
    const float4* ep = (const float4*)(emb + (size_t)k * DIM);
    const float4* xp = (const float4*)(x + rbase);
    float4* op = (float4*)(outq + rbase);
#pragma unroll
    for (int i = 0; i < 16; ++i) {
      const int c4 = q + (i << 2);
      float4 e4 = ep[c4], x4 = xp[c4];
      op[c4] = e4;   // quantized_st value == quantized_x
      float dx = x4.x - e4.x, dy = x4.y - e4.y, dz = x4.z - e4.z, dw = x4.w - e4.w;
      lsum += dx * dx + dy * dy + dz * dz + dw * dw;
    }
  }
  red2[t] = lsum;
  __syncthreads();
  for (int s2 = 128; s2 > 0; s2 >>= 1) {
    if (t < s2) red2[t] += red2[t + s2];
    __syncthreads();
  }
  if (t == 0) partial[blk] = red2[0];
}

// ---------------------------------------------------------------------------
// Kernel 3: deterministic fixed-order final reduction -> two (equal) losses.
// ---------------------------------------------------------------------------
__global__ void vq_finalize_kernel(const float* __restrict__ partial, float* __restrict__ out) {
  if (threadIdx.x == 0 && blockIdx.x == 0) {
    float s = 0.f;
    for (int i = 0; i < NBLKS; ++i) s += partial[i];
    const float loss = s * (1.0f / (float)NDOUT);
    out[NDOUT]     = loss;   // loss_commitment
    out[NDOUT + 1] = loss;   // loss_dictionary (same value, different gradient)
  }
}

extern "C" void kernel_launch(void* const* d_in, const int* in_sizes, int n_in,
                              void* d_out, int out_size, void* d_ws, size_t ws_size,
                              hipStream_t stream) {
  (void)in_sizes; (void)n_in; (void)out_size; (void)ws_size;
  const float* x   = (const float*)d_in[0];     // [32768, 256] fp32
  const float* emb = (const float*)d_in[1];     // [8192, 256] fp32
  float* out = (float*)d_out;                   // [N*D quantized | lc | ld | N mapping]
  float* ws  = (float*)d_ws;
  float* enorm   = ws;            // 8192 floats
  float* partial = ws + KCODES;   // 512 floats

  vq_enorm_kernel<<<KCODES / 8, 256, 0, stream>>>(emb, enorm);
  vq_main_kernel<<<NBLKS, 256, 0, stream>>>(x, emb, enorm,
                                            out, out + NDOUT + 2, partial);
  vq_finalize_kernel<<<1, 32, 0, stream>>>(partial, out);
}
